// SparseMOE_4930622456419
// MI455X (gfx1250) — matrix operand order repaired and checked by reading, not compiled, not run on hardware
//
#include <hip/hip_runtime.h>

typedef __attribute__((ext_vector_type(16))) _Float16 v16h;
typedef __attribute__((ext_vector_type(8)))  _Float16 v8h;
typedef __attribute__((ext_vector_type(4)))  _Float16 v4h;
typedef __attribute__((ext_vector_type(8)))  float    v8f;
typedef __attribute__((ext_vector_type(4)))  float    v4f;

#define B_ 4
#define S_ 2048
#define T_ (B_ * S_)   // 8192 tokens
#define H_ 1024
#define E_ 8
#define K_ 2

// ---------------------------------------------------------------------------
// 1) zero final-output accumulation region + per-expert counters
// ---------------------------------------------------------------------------
__global__ void moe_zero_kernel(float* __restrict__ out, int* __restrict__ counts, long n) {
    long i = (long)blockIdx.x * blockDim.x + threadIdx.x;
    if (i < n) out[i] = 0.0f;
    if (i < E_) counts[i] = 0;
}

// ---------------------------------------------------------------------------
// 2) x: f32 -> f16 conversion (vectorized x4, layout unchanged: [T][H])
// ---------------------------------------------------------------------------
__global__ void moe_cvt_x_kernel(const float* __restrict__ src, _Float16* __restrict__ dst, long n4) {
    long i = (long)blockIdx.x * blockDim.x + threadIdx.x;
    if (i >= n4) return;
    v4f f = ((const v4f*)src)[i];
    v4h h;
    h.x = (_Float16)f.x; h.y = (_Float16)f.y;
    h.z = (_Float16)f.z; h.w = (_Float16)f.w;
    ((v4h*)dst)[i] = h;
}

// ---------------------------------------------------------------------------
// 3) expert_w: f32 -> f16 convert + transpose  W[e][k][o] -> Wt[e][o][k]
//    LDS-tiled 64x64 so both global read and write are coalesced/vectorized.
//    grid = (H/64 [k-tiles], H/64 [o-tiles], E), block = 256
// ---------------------------------------------------------------------------
__global__ __launch_bounds__(256) void moe_cvt_w_kernel(const float* __restrict__ w,
                                                        _Float16* __restrict__ wt)
{
    __shared__ __align__(16) _Float16 tile[64][72];   // 72 = 64 + 8 pad (16B-aligned rows)
    const int e  = blockIdx.z;
    const int k0 = blockIdx.x * 64;
    const int o0 = blockIdx.y * 64;

    // load 64(k) x 64(o) f32 tile, convert, store transposed into LDS
    const float* __restrict__ src = w + ((size_t)e * H_ + k0) * H_ + o0;
    const int r0 = threadIdx.x >> 4;          // 0..15
    const int c4 = (threadIdx.x & 15) * 4;    // 0,4,...,60
#pragma unroll
    for (int rr = 0; rr < 64; rr += 16) {
        const int r = rr + r0;                // local k
        v4f f = *(const v4f*)(src + (size_t)r * H_ + c4);
        tile[c4 + 0][r] = (_Float16)f.x;
        tile[c4 + 1][r] = (_Float16)f.y;
        tile[c4 + 2][r] = (_Float16)f.z;
        tile[c4 + 3][r] = (_Float16)f.w;
    }
    __syncthreads();

    // store 64(o) x 64(k) f16 tile with 16B vector writes (contiguous k)
    _Float16* __restrict__ dst = wt + ((size_t)e * H_ + o0) * H_ + k0;
    const int oo = threadIdx.x >> 3;          // 0..31
    const int k8 = (threadIdx.x & 7) * 8;     // 0,8,...,56
#pragma unroll
    for (int rr = 0; rr < 64; rr += 32) {
        const int o = rr + oo;                // local o
        *(v8h*)(dst + (size_t)o * H_ + k8) = *(const v8h*)&tile[o][k8];
    }
}

// ---------------------------------------------------------------------------
// 4) router: logits -> softmax -> top-2 -> renormalize; build expert lists
// ---------------------------------------------------------------------------
__global__ __launch_bounds__(256) void moe_router_kernel(
    const float* __restrict__ x, const float* __restrict__ gw, const float* __restrict__ gb,
    float* __restrict__ probs_out, int* __restrict__ topk_out,
    int* __restrict__ counts, int* __restrict__ tok_list, float* __restrict__ prob_list)
{
    __shared__ float gws[E_ * H_];   // 32 KB of 320 KB LDS
    for (int i = threadIdx.x; i < E_ * H_; i += blockDim.x) gws[i] = gw[i];
    __syncthreads();

    const int t = blockIdx.x * blockDim.x + threadIdx.x;   // t < T_ by construction
    const float* __restrict__ xr = x + (size_t)t * H_;

    float acc[E_];
#pragma unroll
    for (int e = 0; e < E_; ++e) acc[e] = gb[e];

    for (int h = 0; h < H_; h += 4) {
        v4f xv = *(const v4f*)(xr + h);
#pragma unroll
        for (int e = 0; e < E_; ++e) {
            const float* g = gws + e * H_ + h;
            acc[e] += xv.x * g[0] + xv.y * g[1] + xv.z * g[2] + xv.w * g[3];
        }
    }

    // softmax over E
    float m = acc[0];
#pragma unroll
    for (int e = 1; e < E_; ++e) m = fmaxf(m, acc[e]);
    float p[E_], s = 0.0f;
#pragma unroll
    for (int e = 0; e < E_; ++e) { p[e] = __expf(acc[e] - m); s += p[e]; }
    const float inv = 1.0f / s;
#pragma unroll
    for (int e = 0; e < E_; ++e) { p[e] *= inv; probs_out[(size_t)t * E_ + e] = p[e]; }

    // top-2 (lowest index wins ties, matching lax.top_k)
    int i0 = 0; float p0 = p[0];
#pragma unroll
    for (int e = 1; e < E_; ++e) if (p[e] > p0) { p0 = p[e]; i0 = e; }
    int i1 = (i0 == 0) ? 1 : 0; float p1 = p[i1];
#pragma unroll
    for (int e = 0; e < E_; ++e) if (e != i0 && p[e] > p1) { p1 = p[e]; i1 = e; }

    topk_out[t * K_ + 0] = i0;
    topk_out[t * K_ + 1] = i1;

    const float rs = 1.0f / (p0 + p1);
    int pos0 = atomicAdd(&counts[i0], 1);
    tok_list[i0 * T_ + pos0] = t;  prob_list[i0 * T_ + pos0] = p0 * rs;
    int pos1 = atomicAdd(&counts[i1], 1);
    tok_list[i1 * T_ + pos1] = t;  prob_list[i1 * T_ + pos1] = p1 * rs;
}

// ---------------------------------------------------------------------------
// 5) grouped expert GEMM: gathered tokens x Wt[e], f16 WMMA, f32 accumulate,
//    scaled scatter-add into out.  grid = (H/64, T/32, E), block = 256 (8 waves)
//    Wt is o-major / k-contiguous, so every operand load is a 16B b128 load.
// ---------------------------------------------------------------------------
__global__ __launch_bounds__(256) void moe_gemm_kernel(
    const _Float16* __restrict__ xh, const _Float16* __restrict__ wt,
    const float* __restrict__ eb, const int* __restrict__ counts,
    const int* __restrict__ tok_list, const float* __restrict__ prob_list,
    float* __restrict__ out)
{
    const int e   = blockIdx.z;
    const int cnt = counts[e];
    const int rowBase = blockIdx.y * 32;
    if (rowBase >= cnt) return;          // uniform per block: EXEC stays all-1s

    const int lane = threadIdx.x & 31;
    const int wave = threadIdx.x >> 5;   // 0..7
    const int wr   = (wave >> 2) & 1;    // row sub-tile (2)
    const int wc   = wave & 3;           // col sub-tile (4)
    const int m0   = rowBase + wr * 16;  // first of this wave's 16 token rows
    const int nb   = blockIdx.x * 64 + wc * 16;
    const int n    = lane & 15;          // N index within tile (B/C/D lanes)
    const int hg   = lane >> 4;          // lane half-group

    // A-matrix row gather: lane's M = lane & 15 (both half-groups)
    const int arow = m0 + (lane & 15);
    const int tA   = tok_list[e * T_ + (arow < cnt ? arow : 0)];
    const _Float16* __restrict__ ar = xh + (size_t)tA * H_;
    // B operand: lane n holds column (nb+n) of W = row (nb+n) of Wt, k-contiguous
    const _Float16* __restrict__ wp = wt + ((size_t)e * H_ + nb + n) * H_;

    v8f acc = {};
    for (int kb = 0; kb < H_; kb += 32) {
        // A 16x32 f16 layout: elems 0..7  -> K = kb + i      + 8*hg
        //                     elems 8..15 -> K = kb + 16 + i + 8*hg
        const v8h alo = *(const v8h*)(ar + kb + 8 * hg);
        const v8h ahi = *(const v8h*)(ar + kb + 16 + 8 * hg);
        // B 32x16 f16 layout: same K order as A, N = n
        const v8h blo = *(const v8h*)(wp + kb + 8 * hg);
        const v8h bhi = *(const v8h*)(wp + kb + 16 + 8 * hg);

        v16h a, b;
#pragma unroll
        for (int i = 0; i < 8; ++i) {
            a[i] = alo[i];  a[8 + i] = ahi[i];
            b[i] = blo[i];  b[8 + i] = bhi[i];
        }

        acc = __builtin_amdgcn_wmma_f32_16x16x32_f16(
            /*neg_a=*/false, a, /*neg_b=*/false, b,
            /*c_mod=*/(short)0, acc, /*reuse_a=*/false, /*reuse_b=*/false);
    }

    // epilogue: D VGPR v holds M = v + 8*hg, N = n
    const float bv = eb[e * H_ + nb + n];
#pragma unroll
    for (int v = 0; v < 8; ++v) {
        const int row = m0 + v + 8 * hg;
        if (row < cnt) {
            const int   t = tok_list[e * T_ + row];
            const float p = prob_list[e * T_ + row];
            atomicAdd(out + (size_t)t * H_ + nb + n, p * (acc[v] + bv));
        }
    }
}

// ---------------------------------------------------------------------------
// host launcher
// ---------------------------------------------------------------------------
extern "C" void kernel_launch(void* const* d_in, const int* in_sizes, int n_in,
                              void* d_out, int out_size, void* d_ws, size_t ws_size,
                              hipStream_t stream)
{
    const float* x  = (const float*)d_in[0];   // [B,S,H]
    const float* gw = (const float*)d_in[1];   // [E,H]
    const float* gb = (const float*)d_in[2];   // [E]
    const float* ew = (const float*)d_in[3];   // [E,H,H]  (k-major, o-contiguous)
    const float* eb = (const float*)d_in[4];   // [E,H]

    // outputs concatenated: final_output [T*H] f32, top_k_index [T*K] i32,
    // routing_probs [T*E] f32 (all 4-byte elements)
    float* out_y   = (float*)d_out;
    int*   out_idx = (int*)((char*)d_out + (size_t)T_ * H_ * sizeof(float));
    float* out_p   = (float*)((char*)d_out + ((size_t)T_ * H_ + (size_t)T_ * K_) * sizeof(float));

    // workspace layout
    char* ws = (char*)d_ws;
    _Float16* xh = (_Float16*)ws;                                   // T*H f16
    _Float16* wt = (_Float16*)(ws + (size_t)T_ * H_ * 2);           // E*H*H f16 (transposed)
    size_t off = (size_t)T_ * H_ * 2 + (size_t)E_ * H_ * H_ * 2;
    int* counts = (int*)(ws + off);          off += 16 * sizeof(int);
    int* tok_list = (int*)(ws + off);        off += (size_t)E_ * T_ * sizeof(int);
    float* prob_list = (float*)(ws + off);

    moe_zero_kernel<<<(T_ * H_) / 256, 256, 0, stream>>>(out_y, counts, (long)T_ * H_);
    moe_cvt_x_kernel<<<(T_ * H_ / 4) / 256, 256, 0, stream>>>(x, xh, (long)(T_ * H_ / 4));
    moe_cvt_w_kernel<<<dim3(H_ / 64, H_ / 64, E_), 256, 0, stream>>>(ew, wt);
    moe_router_kernel<<<T_ / 256, 256, 0, stream>>>(x, gw, gb, out_p, out_idx,
                                                    counts, tok_list, prob_list);
    moe_gemm_kernel<<<dim3(H_ / 64, T_ / 32, E_), 256, 0, stream>>>(
        xh, wt, eb, counts, tok_list, prob_list, out_y);
}
